// GraphTransformer_25254407700772
// MI455X (gfx1250) — matrix operand order repaired and checked by reading, not compiled, not run on hardware
//
#include <hip/hip_runtime.h>
#include <hip/hip_bf16.h>

typedef __attribute__((ext_vector_type(16))) __bf16 v16bf;
typedef __attribute__((ext_vector_type(8)))  float  v8f;
typedef int v4i __attribute__((vector_size(16)));   // payload type for async-LDS builtins

#define HEADS   4
#define CDIM    32
#define DIM     128           // IN_DIM == HEADS*CDIM
#define XSTR    132           // fp32 LDS row stride: bank-conflict-free, 16B-aligned rows
#define ALPHA_SCALE 0.17677669529663687f   // 1/sqrt(32)

#define AS1 __attribute__((address_space(1)))
#define AS3 __attribute__((address_space(3)))

#if defined(__gfx1250__) && __has_builtin(__builtin_amdgcn_global_load_async_to_lds_b128)
#define USE_ASYNC_LDS 1
#else
#define USE_ASYNC_LDS 0
#endif

__device__ __forceinline__ void wait_async_zero() {
#if __has_builtin(__builtin_amdgcn_s_wait_asynccnt)
    __builtin_amdgcn_s_wait_asynccnt(0);
#else
    asm volatile("s_wait_asynccnt 0x0" ::: "memory");
#endif
}

__device__ __forceinline__ unsigned short f2bf_u(float f) {
    union { float f; unsigned u; } x; x.f = f;
    unsigned r = x.u + 0x7FFFu + ((x.u >> 16) & 1u);   // round-to-nearest-even
    return (unsigned short)(r >> 16);
}

// ---------------------------------------------------------------------------
// Pack 8 fp32 128x128 weight matrices (row-major [K][N]) into bf16 fragments
// laid out as V_WMMA_F32_16X16X32_BF16 B operands, contiguous k order (same as A):
//   fragment(m, nt, kc): lane l, elem i -> K = kc*32 + ((l>=16)?16:0) + i,
//                                          N = nt*16 + (l&15)
// Flat element index = ((m*8+nt)*4+kc)*512 + l*16 + i  (contiguous per lane)
// ---------------------------------------------------------------------------
__global__ void pack_weights(const float* __restrict__ W0, const float* __restrict__ W1,
                             const float* __restrict__ W2, const float* __restrict__ W3,
                             const float* __restrict__ W4, const float* __restrict__ W5,
                             const float* __restrict__ W6, const float* __restrict__ W7,
                             unsigned short* __restrict__ pw) {
    int t  = blockIdx.x * blockDim.x + threadIdx.x;    // 0 .. 131071
    int i  = t & 15;
    int l  = (t >> 4) & 31;
    int kc = (t >> 9) & 3;
    int nt = (t >> 11) & 7;
    int m  = t >> 14;
    const float* W;
    switch (m) {
        case 0: W = W0; break; case 1: W = W1; break;
        case 2: W = W2; break; case 3: W = W3; break;
        case 4: W = W4; break; case 5: W = W5; break;
        case 6: W = W6; break; default: W = W7; break;
    }
    int K = kc * 32 + ((l >> 4) << 4) + i;
    int Nn = nt * 16 + (l & 15);
    pw[t] = f2bf_u(W[K * DIM + Nn]);
}

// ---------------------------------------------------------------------------
// Fused Q/K/V/Skip projection: out_m = x @ W_m + b_m for m in {q,k,v,s}.
// Block = 256 threads (8 waves), 16 rows per block; wave w = column tile w.
// x tile staged fp32 via GLOBAL_LOAD_ASYNC_TO_LDS_B128 (ASYNCcnt), converted
// to bf16 during fragment build. Four independent WMMA accumulator chains
// (one per output matrix) are interleaved over K to avoid D->C RAW bubbles.
// ---------------------------------------------------------------------------
__global__ void __launch_bounds__(256) gemm_qkvs(
        const float* __restrict__ xin,
        const unsigned short* __restrict__ pw,     // 4 packed matrices (65536 elems)
        const float* __restrict__ bq, const float* __restrict__ bk,
        const float* __restrict__ bv, const float* __restrict__ bs,
        float* __restrict__ qo, float* __restrict__ ko,
        float* __restrict__ vo, float* __restrict__ so,
        int N) {
    __shared__ float xtf[16 * XSTR];

    int tid  = threadIdx.x;
    int row0 = blockIdx.x * 16;
    bool full = (row0 + 16) <= N;

#if USE_ASYNC_LDS
    if (full) {
        // 16 rows x 128 fp32 = 512 16-byte chunks; 2 async copies per thread
        for (int t2 = tid; t2 < 512; t2 += 256) {
            int r  = t2 >> 5;
            int c4 = (t2 & 31) << 2;
            __builtin_amdgcn_global_load_async_to_lds_b128(
                (AS1 v4i*)(xin + (size_t)(row0 + r) * DIM + c4),
                (AS3 v4i*)(&xtf[r * XSTR + c4]), 0, 0);
        }
        wait_async_zero();
    } else
#endif
    {
        for (int idx = tid; idx < 16 * DIM; idx += 256) {
            int r = idx >> 7, c = idx & 127;
            int gr = row0 + r;
            xtf[r * XSTR + c] = (gr < N) ? xin[(size_t)gr * DIM + c] : 0.0f;
        }
    }
    __syncthreads();

    int wave = tid >> 5;            // col tile 0..7
    int lane = tid & 31;
    int mrow = lane & 15;
    int hi8  = (lane >> 4) << 4;    // +16 K-offset for lanes 16..31 (A layout)

    // Build A fragments for all 4 K-chunks (shared across the 4 matrices)
    v16bf a[4];
#pragma unroll
    for (int kc = 0; kc < 4; ++kc) {
#pragma unroll
        for (int i = 0; i < 16; ++i) {
            int K = kc * 32 + i + hi8;
            a[kc][i] = (__bf16)xtf[mrow * XSTR + K];
        }
    }

    // Four independent accumulator chains interleaved over K
    v8f c[4] = { {}, {}, {}, {} };
#pragma unroll
    for (int kc = 0; kc < 4; ++kc) {
#pragma unroll
        for (int m = 0; m < 4; ++m) {
            const v16bf b = *(const v16bf*)(pw + (size_t)(((m * 8 + wave) * 4 + kc) * 512 + lane * 16));
            c[m] = __builtin_amdgcn_wmma_f32_16x16x32_bf16(
                       false, a[kc], false, b, (short)0, c[m], false, false);
        }
    }

    const float* biases[4] = { bq, bk, bv, bs };
    float*       outs[4]   = { qo, ko, vo, so };
    int ncol  = wave * 16 + (lane & 15);
    int rbase = row0 + ((lane >> 4) << 3);

    if (full) {
#pragma unroll
        for (int m = 0; m < 4; ++m) {
            float bias = biases[m][ncol];
            float* o = outs[m];
#pragma unroll
            for (int r = 0; r < 8; ++r)
                o[(size_t)(rbase + r) * DIM + ncol] = c[m][r] + bias;
        }
    } else {
#pragma unroll
        for (int m = 0; m < 4; ++m) {
            float bias = biases[m][ncol];
            float* o = outs[m];
#pragma unroll
            for (int r = 0; r < 8; ++r) {
                int gr = rbase + r;
                if (gr < N) o[(size_t)gr * DIM + ncol] = c[m][r] + bias;
            }
        }
    }
}

// ---------------------------------------------------------------------------
__global__ void init_nodes(unsigned* __restrict__ amax, float* __restrict__ denom, int n4) {
    int i = blockIdx.x * blockDim.x + threadIdx.x;
    if (i < n4) { amax[i] = 0u; denom[i] = 0.0f; }
}

// Map float -> monotonically ordered uint (for atomicMax-based segment max)
__device__ __forceinline__ unsigned f2ord(float f) {
    unsigned u = __float_as_uint(f);
    return u ^ (0x80000000u | (unsigned)((int)u >> 31));
}
__device__ __forceinline__ float ord2f(unsigned u) {
    u = (u & 0x80000000u) ? (u ^ 0x80000000u) : ~u;
    return __uint_as_float(u);
}

// alpha[e,h] = <q[dst,h,:], k[src,h,:] + ew*We[h,:]> / sqrt(C);  segment max via atomicMax
__global__ void edge_alpha(const float* __restrict__ q, const float* __restrict__ k,
                           const float* __restrict__ We, const float* __restrict__ ew,
                           const int* __restrict__ ei, unsigned* __restrict__ amax,
                           float* __restrict__ alpha, int E) {
    int idx = blockIdx.x * blockDim.x + threadIdx.x;
    if (idx >= E * HEADS) return;
    int e = idx >> 2, h = idx & 3;
    int s = ei[e], d = ei[E + e];
    float w = ew[e];
    const float4* qp = (const float4*)(q + (size_t)d * DIM + h * CDIM);
    const float4* kp = (const float4*)(k + (size_t)s * DIM + h * CDIM);
    const float4* wp = (const float4*)(We + h * CDIM);
    float acc = 0.0f;
#pragma unroll
    for (int j = 0; j < 8; ++j) {
        float4 qv = qp[j], kv = kp[j], wv = wp[j];
        acc += qv.x * (kv.x + w * wv.x);
        acc += qv.y * (kv.y + w * wv.y);
        acc += qv.z * (kv.z + w * wv.z);
        acc += qv.w * (kv.w + w * wv.w);
    }
    acc *= ALPHA_SCALE;
    alpha[idx] = acc;
    atomicMax(&amax[(size_t)d * HEADS + h], f2ord(acc));
}

// p = exp(alpha - amax[dst]); denom[dst] += p   (alpha overwritten with p)
__global__ void edge_softmax(const unsigned* __restrict__ amax, const int* __restrict__ ei,
                             float* __restrict__ alpha, float* __restrict__ denom, int E) {
    int idx = blockIdx.x * blockDim.x + threadIdx.x;
    if (idx >= E * HEADS) return;
    int e = idx >> 2, h = idx & 3;
    int d = ei[E + e];
    float m = ord2f(amax[(size_t)d * HEADS + h]);
    float p = __expf(alpha[idx] - m);
    alpha[idx] = p;
    atomicAdd(&denom[(size_t)d * HEADS + h], p);
}

// acc[dst,i] += (v[src,i] + ew*We[i]) * p/denom[dst,h]   (i = 0..127, h = i/32)
__global__ void edge_message(const float* __restrict__ v, const float* __restrict__ We,
                             const float* __restrict__ ew, const int* __restrict__ ei,
                             const float* __restrict__ alpha, const float* __restrict__ denom,
                             float* __restrict__ acc, int E) {
    int idx = blockIdx.x * blockDim.x + threadIdx.x;
    if (idx >= E * DIM) return;
    int e = idx >> 7, i = idx & 127;
    int h = i >> 5;
    int s = ei[e], d = ei[E + e];
    float p  = alpha[(size_t)e * HEADS + h];
    float dn = denom[(size_t)d * HEADS + h];
    float wgt = p / dn;
    float msg = (v[(size_t)s * DIM + i] + ew[e] * We[i]) * wgt;
    atomicAdd(&acc[(size_t)d * DIM + i], msg);
}

// ---------------------------------------------------------------------------
extern "C" void kernel_launch(void* const* d_in, const int* in_sizes, int n_in,
                              void* d_out, int out_size, void* d_ws, size_t ws_size,
                              hipStream_t stream) {
    const float* x  = (const float*)d_in[0];
    const float* ew = (const float*)d_in[1];
    const int*   ei = (const int*)d_in[20];
    int N = in_sizes[0] / DIM;
    int E = in_sizes[1];

    // per-layer params: [layer][Wq,bq,Wk,bk,Wv,bv,We,Ws,bs] starting at idx 2 and 11
    const float* P[2][9];
    for (int l = 0; l < 2; ++l)
        for (int j = 0; j < 9; ++j)
            P[l][j] = (const float*)d_in[2 + l * 9 + j];

    char* ws = (char*)d_ws;
    size_t off = 0;
    unsigned short* pw = (unsigned short*)(ws + off); off += 8 * 16384 * sizeof(unsigned short);
    size_t nfeat = (size_t)N * DIM * sizeof(float);
    float* qb  = (float*)(ws + off); off += nfeat;
    float* kb  = (float*)(ws + off); off += nfeat;
    float* vb  = (float*)(ws + off); off += nfeat;
    float* hb  = (float*)(ws + off); off += nfeat;                 // layer-1 output
    float* alpha = (float*)(ws + off); off += (size_t)E * HEADS * sizeof(float);
    unsigned* amax = (unsigned*)(ws + off); off += (size_t)N * HEADS * sizeof(unsigned);
    float* denom = (float*)(ws + off); off += (size_t)N * HEADS * sizeof(float);

    // pack all 8 weight matrices to WMMA-ready bf16 fragments
    pack_weights<<<512, 256, 0, stream>>>(
        P[0][0], P[0][2], P[0][4], P[0][7],     // Wq1, Wk1, Wv1, Ws1
        P[1][0], P[1][2], P[1][4], P[1][7],     // Wq2, Wk2, Wv2, Ws2
        pw);

    int gemm_blocks = (N + 15) / 16;
    int eh_blocks   = (E * HEADS + 255) / 256;
    int em_blocks   = (E * DIM + 255) / 256;
    int in_blocks   = (N * HEADS + 255) / 256;

    for (int layer = 0; layer < 2; ++layer) {
        const float* xin  = layer ? hb : x;
        float* accout     = layer ? (float*)d_out : hb;
        const float* We_  = P[layer][6];
        const unsigned short* pwl = pw + (size_t)layer * 4 * 16384;

        init_nodes<<<in_blocks, 256, 0, stream>>>(amax, denom, N * HEADS);

        // q,k,v and acc=x@Ws+bs (skip term pre-seeds the aggregation buffer)
        gemm_qkvs<<<gemm_blocks, 256, 0, stream>>>(
            xin, pwl,
            P[layer][1], P[layer][3], P[layer][5], P[layer][8],   // bq,bk,bv,bs
            qb, kb, vb, accout, N);

        edge_alpha<<<eh_blocks, 256, 0, stream>>>(qb, kb, We_, ew, ei, amax, alpha, E);
        edge_softmax<<<eh_blocks, 256, 0, stream>>>(amax, ei, alpha, denom, E);
        edge_message<<<em_blocks, 256, 0, stream>>>(vb, We_, ew, ei, alpha, denom, accout, E);
    }
}
